// RNN_4561255269000
// MI455X (gfx1250) — matrix operand order repaired and checked by reading, not compiled, not run on hardware
//
#include <hip/hip_runtime.h>

// CDNA5 / gfx1250, wave32. Matrix ops via v_wmma_f32_16x16x32_f16.

typedef __attribute__((ext_vector_type(16))) _Float16 v16h;
typedef __attribute__((ext_vector_type(8)))  _Float16 v8h;
typedef __attribute__((ext_vector_type(8)))  float    v8f;

#define HIDDEN 128
#define INP    28
#define T_LEN  256
#define OUTD   784   // 28*28
#define BATCH  4096

__device__ __forceinline__ float4 ldg4(const float* p) { return *(const float4*)p; }

__device__ __forceinline__ void put4(v16h& d, int e, float4 v) {
    d[e + 0] = (_Float16)v.x;
    d[e + 1] = (_Float16)v.y;
    d[e + 2] = (_Float16)v.z;
    d[e + 3] = (_Float16)v.w;
}

__device__ __forceinline__ v8f wmma_f16(v16h a, v16h b, v8f c) {
    // D = A(16x32 f16) * B(32x16 f16) + C(16x16 f32)
    return __builtin_amdgcn_wmma_f32_16x16x32_f16(
        /*neg_a=*/false, a, /*neg_b=*/false, b,
        /*c_mod=*/(short)0, c, /*reuse_a=*/false, /*reuse_b=*/false);
}

// ---------------------------------------------------------------------------
// Kernel 1: recurrent scan.
//   grid = 256 blocks (one per 16 batch rows), block = 128 threads = 4 waves.
//   Wave w owns hidden columns [w*32, w*32+32) (2 WMMA N-tiles).
//   h kept in double-buffered LDS (f16); W_ih/W_hh fragments in VGPRs.
//   Bias pre-folded into the WMMA C-operand; relu done in f16 after the
//   packed f32->f16 conversion (cvt_pk + pk_max instead of canon+max+cvt).
// A-frag layout (16-bit A 16x32): lanes 0-15 = rows, elems 0-15 -> K ks..ks+15,
//   ks = (lane<16 ? 0 : 16), same K order as B.
// B-frag layout (16-bit B 32x16): lane = column n, elems 0-15 -> K ks..ks+15,
//   ks = (lane<16 ? 0 : 16).
// C/D layout: vgpr r = row (lane<16 ? r : r+8), col = lane&15.
// ---------------------------------------------------------------------------
__global__ __launch_bounds__(128)
void rnn_scan_kernel(const float* __restrict__ x,     // [B, T, IN]
                     const float* __restrict__ h0,    // [B, HIDDEN]
                     const float* __restrict__ W_ih,  // [HIDDEN, IN]
                     const float* __restrict__ b_ih,  // [HIDDEN]
                     const float* __restrict__ W_hh,  // [HIDDEN, HIDDEN]
                     const float* __restrict__ b_hh,  // [HIDDEN]
                     float* __restrict__ hid_out)     // [B, HIDDEN] f32
{
    __shared__ __align__(16) _Float16 hbuf[2][16][HIDDEN + 8]; // padded rows

    const int lane = threadIdx.x & 31;
    const int wave = threadIdx.x >> 5;
    const int m    = lane & 15;
    const int hiH  = (lane & 16) ? 1 : 0;
    const int lo   = hiH * 8;     // unused; A-frag K offset is ks0
    const int ks0  = hiH * 16;    // B-frag K offset
    const int b0   = blockIdx.x * 16;
    const int n0w  = wave * 32;

    // ---- preload weight B-fragments + broadcast bias into registers ----
    v16h whh[2][4];
    v16h wih[2];
    v8f  biasv[2];                // bias broadcast into C-operand layout
#pragma unroll
    for (int j = 0; j < 2; ++j) {
        const int n = n0w + j * 16 + m;
        const float bsum = b_ih[n] + b_hh[n];  // same column for all 8 C rows
#pragma unroll
        for (int r = 0; r < 8; ++r) biasv[j][r] = bsum;

        const float* wr = W_hh + (size_t)n * HIDDEN;
#pragma unroll
        for (int c = 0; c < 4; ++c) {
            const int ks = c * 32 + ks0;
#pragma unroll
            for (int q = 0; q < 4; ++q)
                put4(whh[j][c], q * 4, ldg4(wr + ks + q * 4));
        }

        // W_ih: K = 28 padded to 32 (pad lanes' top elements with zero)
        const float* wi = W_ih + (size_t)n * INP;
        v16h wf;
        if (!hiH) {               // K = 0..15, all valid
#pragma unroll
            for (int q = 0; q < 4; ++q)
                put4(wf, q * 4, ldg4(wi + q * 4));
        } else {                  // K = 16..31; 28..31 are zero padding
#pragma unroll
            for (int q = 0; q < 3; ++q)
                put4(wf, q * 4, ldg4(wi + 16 + q * 4));
            put4(wf, 12, make_float4(0.f, 0.f, 0.f, 0.f));
        }
        wih[j] = wf;
    }

    // ---- initialize h buffer 0 from h0 (f32 -> f16) ----
#pragma unroll 4
    for (int r = 0; r < 16; ++r)
        hbuf[0][r][threadIdx.x] =
            (_Float16)h0[(size_t)(b0 + r) * HIDDEN + threadIdx.x];
    __syncthreads();

    const float* xrow = x + (size_t)(b0 + m) * T_LEN * INP;
    int cur = 0;

    for (int t = 0; t < T_LEN; ++t) {
        // ---- x_t A-fragment (K = 28 padded to 32), guarded against OOB ----
        const float* xr = xrow + (size_t)t * INP;
        float4 a0 = ldg4(xr + ks0);           // K ks0..ks0+3
        float4 a1 = ldg4(xr + ks0 + 4);       // K ks0+4..ks0+7
        float4 a2 = ldg4(xr + ks0 + 8);       // K ks0+8..ks0+11 (hi: 24..27)
        float4 a3 = hiH ? make_float4(0.f, 0.f, 0.f, 0.f)   // K 28..31 -> 0
                        : ldg4(xr + 12);                    // K 12..15
        v16h xa;
        put4(xa, 0, a0); put4(xa, 4, a1); put4(xa, 8, a2); put4(xa, 12, a3);

        // Prefetch next step's x row while this step computes.
        __builtin_prefetch(xr + INP, 0, 3);

        // ---- input projection, C pre-seeded with bias ----
        v8f acc[2];
#pragma unroll
        for (int j = 0; j < 2; ++j)
            acc[j] = wmma_f16(xa, wih[j], biasv[j]);

        // ---- gather h A-fragments from LDS (ds_load_b128 pairs) ----
        v16h ha[4];
#pragma unroll
        for (int c = 0; c < 4; ++c) {
            v8h p0 = *(const v8h*)&hbuf[cur][m][c * 32 + ks0];
            v8h p1 = *(const v8h*)&hbuf[cur][m][c * 32 + ks0 + 8];
#pragma unroll
            for (int e = 0; e < 8; ++e) { ha[c][e] = p0[e]; ha[c][8 + e] = p1[e]; }
        }

        // ---- recurrent matmul: 8 chained WMMAs (2 N-tiles x 4 K-chunks) ----
#pragma unroll
        for (int j = 0; j < 2; ++j)
#pragma unroll
            for (int c = 0; c < 4; ++c)
                acc[j] = wmma_f16(ha[c], whh[j][c], acc[j]);

        // ---- convert to f16 (cvt_pk pairs), relu in f16, scatter-store ----
        const int nxt = cur ^ 1;
#pragma unroll
        for (int j = 0; j < 2; ++j) {
            const int n = n0w + j * 16 + m;
            v8h hv;
#pragma unroll
            for (int r = 0; r < 8; ++r) hv[r] = (_Float16)acc[j][r];
            const _Float16 zz = (_Float16)0.f;
#pragma unroll
            for (int r = 0; r < 8; ++r) hv[r] = hv[r] > zz ? hv[r] : zz;
#pragma unroll
            for (int r = 0; r < 8; ++r)
                hbuf[nxt][r + hiH * 8][n] = hv[r];
        }
        __syncthreads();   // single barrier per step (double-buffered h)
        cur = nxt;
    }

    // ---- final hidden state: coalesced read-back from the live h buffer ----
#pragma unroll 4
    for (int r = 0; r < 16; ++r)
        hid_out[(size_t)(b0 + r) * HIDDEN + threadIdx.x] =
            (float)hbuf[cur][r][threadIdx.x];
}

// ---------------------------------------------------------------------------
// Kernel 2: out = h_final @ W_ho^T + b_ho   ([4096,128] x [128,784])
//   1792 waves: each wave = one 16-row tile x 7 column tiles (A reused).
// ---------------------------------------------------------------------------
__global__ __launch_bounds__(256)
void rnn_out_kernel(const float* __restrict__ hid,   // [B, HIDDEN] f32
                    const float* __restrict__ W_ho,  // [OUTD, HIDDEN]
                    const float* __restrict__ b_ho,  // [OUTD]
                    float* __restrict__ out)         // [B, OUTD]
{
    const int lane = threadIdx.x & 31;
    const int wave = threadIdx.x >> 5;
    const int gid  = blockIdx.x * 8 + wave;     // 0..1791
    const int mt   = gid / 7;                   // 0..255 row tile
    const int ng   = gid % 7;                   // column group (7 tiles each)
    const int b0   = mt * 16;
    const int m    = lane & 15;
    const int hiH  = (lane & 16) ? 1 : 0;
    const int lo   = hiH * 8;
    const int ks0  = hiH * 16;

    // A-fragments: h rows b0..b0+15 (f32 -> f16), contiguous A layout
    v16h ha[4];
    const float* hr = hid + (size_t)(b0 + m) * HIDDEN;
#pragma unroll
    for (int c = 0; c < 4; ++c) {
        put4(ha[c], 0,  ldg4(hr + c * 32 + ks0));
        put4(ha[c], 4,  ldg4(hr + c * 32 + ks0 + 4));
        put4(ha[c], 8,  ldg4(hr + c * 32 + ks0 + 8));
        put4(ha[c], 12, ldg4(hr + c * 32 + ks0 + 12));
    }

    for (int u = 0; u < 7; ++u) {
        const int n = (ng * 7 + u) * 16 + m;    // this lane's output column
        const float* wr = W_ho + (size_t)n * HIDDEN;
        v8f acc = {};
#pragma unroll
        for (int c = 0; c < 4; ++c) {
            v16h bf;
            const int ks = c * 32 + ks0;
#pragma unroll
            for (int q = 0; q < 4; ++q)
                put4(bf, q * 4, ldg4(wr + ks + q * 4));
            acc = wmma_f16(ha[c], bf, acc);
        }
        const float bb = b_ho[n];
#pragma unroll
        for (int r = 0; r < 8; ++r)
            out[(size_t)(b0 + r + hiH * 8) * OUTD + n] = acc[r] + bb;
    }
}

// ---------------------------------------------------------------------------
extern "C" void kernel_launch(void* const* d_in, const int* in_sizes, int n_in,
                              void* d_out, int out_size, void* d_ws, size_t ws_size,
                              hipStream_t stream) {
    const float* x   = (const float*)d_in[0];
    const float* h0  = (const float*)d_in[1];
    const float* Wih = (const float*)d_in[2];
    const float* bih = (const float*)d_in[3];
    const float* Whh = (const float*)d_in[4];
    const float* bhh = (const float*)d_in[5];
    const float* Who = (const float*)d_in[6];
    const float* bho = (const float*)d_in[7];

    float* out = (float*)d_out;                       // [4096, 784]
    float* hid = out + (size_t)BATCH * OUTD;          // [4096, 128] (2nd output)

    rnn_scan_kernel<<<BATCH / 16, 128, 0, stream>>>(x, h0, Wih, bih, Whh, bhh, hid);
    rnn_out_kernel<<<(BATCH / 16) * 7 / 8, 256, 0, stream>>>(hid, Who, bho, out);
}
